// factor_updating_structure_45664092291697
// MI455X (gfx1250) — compile-verified, matrix operand order checked by reading, not run on hardware
//
#include <hip/hip_runtime.h>
#include <stdint.h>
#include <stddef.h>

// ---------------------------------------------------------------------------
// Problem constants (from reference)
// ---------------------------------------------------------------------------
#define NO_   512
#define NR_   1024
#define DHO_  512
#define DHR_  512
#define DM_   256
#define HW_   49
#define NRHW_ (NR_ * HW_)      // 50176
#define SCALE_ 0.0625f         // 1/sqrt(256 + 1e-10)
#define NEG_  (-1e30f)

typedef __bf16 bf16;
typedef bf16  v16bf __attribute__((ext_vector_type(16)));
typedef bf16  v8bf  __attribute__((ext_vector_type(8)));
typedef float v8f   __attribute__((ext_vector_type(8)));

// float -> bf16, round-to-nearest-even (manual: no reliance on __bf16 arithmetic)
__device__ __forceinline__ bf16 f2bf(float f) {
  union { float f; uint32_t u; } in; in.f = f;
  uint32_t u = in.u;
  uint32_t r = u + 0x7FFFu + ((u >> 16) & 1u);
  union { uint16_t s; bf16 b; } out; out.s = (uint16_t)(r >> 16);
  return out.b;
}

// ---------------------------------------------------------------------------
// bf16 WMMA GEMM:  C[M,N] = A[M,K] * Bt[N,K]^T   (both K-contiguous)
//
// Block tile 128x128, BK=32, 256 threads = 8 waves; each wave owns a 64x32
// sub-tile = 4x2 WMMA(16x16x32) tiles -> 8 v_wmma per K-step, 12 ds_load_b128.
// LDS is ping-pong double buffered: global loads for tile k+1 overlap WMMA on
// tile k; one workgroup barrier per K-step.
//
// Epilogue (compile-time):
//   MODE 0: outF[m*N+n] = C (+bias[n]) (+resid[m*N+n])      fp32 row-major
//   MODE 1: outB[m*N+n] = bf16( relu?(C (+bias[n])) )       bf16 row-major
//   MODE 2: m=(r*49+hw); off=r*N*49+n*49+hw;
//           outF[off] = C + bias[n] + resid[off]            fp32 region layout
// Requires: M%128==0, N%128==0, K%32==0 (true for every call here).
// ---------------------------------------------------------------------------
#define BM 128
#define BN 128
#define BK 32
#define LDK 40   // padded LDS row stride in bf16 elems (80B rows)

template <int MODE, bool RELU, bool HASBIAS, bool HASRES>
__global__ __launch_bounds__(256)
void gemm_bf16_nt(const bf16* __restrict__ A, const bf16* __restrict__ Bt,
                  int M, int N, int K,
                  float* __restrict__ outF, bf16* __restrict__ outB,
                  const float* __restrict__ bias,
                  const float* __restrict__ resid)
{
  __shared__ __align__(16) bf16 sA[2][BM * LDK];
  __shared__ __align__(16) bf16 sB[2][BN * LDK];

  const int tid  = threadIdx.x;
  const int lane = tid & 31;
  const int wave = tid >> 5;
  const int wm   = wave & 1;            // 2 row-slices of 64
  const int wn   = wave >> 1;           // 4 col-slices of 32
  const int r16  = lane & 15;
  const int ksel = (lane >> 4) << 3;    // 0 or 8 (CDNA5 16-bit fragment K split)

  const int bm = blockIdx.x * BM;
  const int bn = blockIdx.y * BN;

  // global-load chunk mapping: 128x32 bf16 = 512 x16B chunks, 2 per thread
  int rowL[2], kcL[2];
  #pragma unroll
  for (int p = 0; p < 2; ++p) {
    int ci = p * 256 + tid;
    rowL[p] = ci >> 2;
    kcL[p]  = ci & 3;
  }

  v8f acc[4][2];
  #pragma unroll
  for (int i = 0; i < 4; ++i)
    #pragma unroll
    for (int j = 0; j < 2; ++j)
      acc[i][j] = (v8f){0.f,0.f,0.f,0.f,0.f,0.f,0.f,0.f};

  v8bf rA[2], rB[2];

  // prologue: fetch K-tile 0
  #pragma unroll
  for (int p = 0; p < 2; ++p) {
    rA[p] = *(const v8bf*)(A  + (size_t)(bm + rowL[p]) * (size_t)K + (size_t)(kcL[p] * 8));
    rB[p] = *(const v8bf*)(Bt + (size_t)(bn + rowL[p]) * (size_t)K + (size_t)(kcL[p] * 8));
  }
  #pragma unroll
  for (int p = 0; p < 2; ++p) {
    *(v8bf*)(&sA[0][rowL[p] * LDK + kcL[p] * 8]) = rA[p];
    *(v8bf*)(&sB[0][rowL[p] * LDK + kcL[p] * 8]) = rB[p];
  }
  __syncthreads();

  const int nk = K / BK;
  for (int kt = 0; kt < nk; ++kt) {
    const int cur = kt & 1;
    const int nxt = cur ^ 1;
    const bool more = (kt + 1) < nk;

    // fetch K-tile kt+1 into registers (overlaps WMMA below)
    if (more) {
      int k0 = (kt + 1) * BK;
      #pragma unroll
      for (int p = 0; p < 2; ++p) {
        const bf16* pa = A  + (size_t)(bm + rowL[p]) * (size_t)K + (size_t)(k0 + kcL[p] * 8);
        const bf16* pb = Bt + (size_t)(bn + rowL[p]) * (size_t)K + (size_t)(k0 + kcL[p] * 8);
        rA[p] = *(const v8bf*)pa;
        rB[p] = *(const v8bf*)pb;
        // warm L2 two tiles ahead (global_prefetch_b8)
        __builtin_prefetch(pa + BK, 0, 1);
        __builtin_prefetch(pb + BK, 0, 1);
      }
    }

    // fragments from current LDS buffer
    v16bf aF[4], bF[2];
    #pragma unroll
    for (int t = 0; t < 4; ++t) {
      const bf16* p = &sA[cur][(wm * 64 + t * 16 + r16) * LDK + ksel];
      v8bf lo = *(const v8bf*)p;
      v8bf hi = *(const v8bf*)(p + 16);
      aF[t] = __builtin_shufflevector(lo, hi, 0,1,2,3,4,5,6,7,8,9,10,11,12,13,14,15);
    }
    #pragma unroll
    for (int t = 0; t < 2; ++t) {
      const bf16* p = &sB[cur][(wn * 32 + t * 16 + r16) * LDK + ksel];
      v8bf lo = *(const v8bf*)p;
      v8bf hi = *(const v8bf*)(p + 16);
      bF[t] = __builtin_shufflevector(lo, hi, 0,1,2,3,4,5,6,7,8,9,10,11,12,13,14,15);
    }

    #pragma unroll
    for (int tm = 0; tm < 4; ++tm)
      #pragma unroll
      for (int tn = 0; tn < 2; ++tn)
        acc[tm][tn] = __builtin_amdgcn_wmma_f32_16x16x32_bf16(
            false, aF[tm], false, bF[tn], (short)0, acc[tm][tn], false, false);

    // stage K-tile kt+1 into the other LDS buffer; that buffer's readers
    // finished before the barrier that ended iteration kt-1.
    if (more) {
      #pragma unroll
      for (int p = 0; p < 2; ++p) {
        *(v8bf*)(&sA[nxt][rowL[p] * LDK + kcL[p] * 8]) = rA[p];
        *(v8bf*)(&sB[nxt][rowL[p] * LDK + kcL[p] * 8]) = rB[p];
      }
    }
    __syncthreads();
  }

  // Epilogue (fully static)
  #pragma unroll
  for (int tm = 0; tm < 4; ++tm) {
    #pragma unroll
    for (int tn = 0; tn < 2; ++tn) {
      int col     = bn + wn * 32 + tn * 16 + r16;
      int rowBase = bm + wm * 64 + tm * 16 + ((lane >> 4) << 3);
      float bv = HASBIAS ? bias[col] : 0.f;
      #pragma unroll
      for (int v = 0; v < 8; ++v) {
        int   m = rowBase + v;
        float c = acc[tm][tn][v] + bv;
        if (RELU) c = fmaxf(c, 0.f);
        if (MODE == 0) {
          size_t off = (size_t)m * (size_t)N + (size_t)col;
          if (HASRES) c += resid[off];
          outF[off] = c;
        } else if (MODE == 1) {
          outB[(size_t)m * (size_t)N + (size_t)col] = f2bf(c);
        } else {
          int r = m / HW_, hw = m - r * HW_;
          size_t off = (size_t)r * (size_t)N * HW_ + (size_t)col * HW_ + (size_t)hw;
          if (HASRES) c += resid[off];
          outF[off] = c;
        }
      }
    }
  }
}

// ---------------------------------------------------------------------------
// Block reductions (256 threads)
// ---------------------------------------------------------------------------
__device__ __forceinline__ float block_max(float v, float* sm) {
  sm[threadIdx.x] = v; __syncthreads();
  for (int s = 128; s > 0; s >>= 1) {
    if (threadIdx.x < s) sm[threadIdx.x] = fmaxf(sm[threadIdx.x], sm[threadIdx.x + s]);
    __syncthreads();
  }
  float r = sm[0]; __syncthreads();
  return r;
}
__device__ __forceinline__ float block_sum(float v, float* sm) {
  sm[threadIdx.x] = v; __syncthreads();
  for (int s = 128; s > 0; s >>= 1) {
    if (threadIdx.x < s) sm[threadIdx.x] += sm[threadIdx.x + s];
    __syncthreads();
  }
  float r = sm[0]; __syncthreads();
  return r;
}

// ---------------------------------------------------------------------------
// Softmax over all 50176 (region,hw) entries per object, masked+scaled
// ---------------------------------------------------------------------------
__global__ __launch_bounds__(256)
void softmax_obj_kernel(const float* __restrict__ sim, const int* __restrict__ mat_object,
                        bf16* __restrict__ prob)
{
  __shared__ float red[256];
  int o = blockIdx.x;
  const float* row  = sim + (size_t)o * NRHW_;
  const int*   mrow = mat_object + o * NR_;

  float mx = -3.0e38f;
  for (int j = threadIdx.x; j < NRHW_; j += 256) {
    float v = (mrow[j / HW_] > 0) ? row[j] * SCALE_ : NEG_;
    mx = fmaxf(mx, v);
  }
  mx = block_max(mx, red);

  float s = 0.f;
  for (int j = threadIdx.x; j < NRHW_; j += 256) {
    float v = (mrow[j / HW_] > 0) ? row[j] * SCALE_ : NEG_;
    s += __expf(v - mx);
  }
  s = block_sum(s, red);
  float inv = 1.0f / s;

  for (int j = threadIdx.x; j < NRHW_; j += 256) {
    float v = (mrow[j / HW_] > 0) ? row[j] * SCALE_ : NEG_;
    prob[(size_t)o * NRHW_ + j] = f2bf(__expf(v - mx) * inv);
  }
}

// ---------------------------------------------------------------------------
// Softmax over 512 objects per (region,hw) row, masked+scaled
// ---------------------------------------------------------------------------
__global__ __launch_bounds__(256)
void softmax_reg_kernel(const float* __restrict__ sim2, const int* __restrict__ mat_region,
                        bf16* __restrict__ prob2)
{
  __shared__ float red[256];
  int j = blockIdx.x;
  int r = j / HW_;
  const float* row  = sim2 + (size_t)j * NO_;
  const int*   mrow = mat_region + r * NO_;
  int i0 = threadIdx.x, i1 = threadIdx.x + 256;

  float v0 = (mrow[i0] > 0) ? row[i0] * SCALE_ : NEG_;
  float v1 = (mrow[i1] > 0) ? row[i1] * SCALE_ : NEG_;
  float mx = block_max(fmaxf(v0, v1), red);
  float e0 = __expf(v0 - mx), e1 = __expf(v1 - mx);
  float s  = block_sum(e0 + e1, red);
  float inv = 1.0f / s;
  prob2[(size_t)j * NO_ + i0] = f2bf(e0 * inv);
  prob2[(size_t)j * NO_ + i1] = f2bf(e1 * inv);
}

// ---------------------------------------------------------------------------
// Packing / conversion kernels
// ---------------------------------------------------------------------------
// feature_region [r, c, hw] ->  frT_relu[(r*49+hw), c] = bf16(relu(x))
//                               frR_cb  [c, (r*49+hw)] = bf16(x)
__global__ __launch_bounds__(256)
void region_pack_kernel(const float* __restrict__ fr, bf16* __restrict__ frT_relu,
                        bf16* __restrict__ frR_cb)
{
  size_t i = (size_t)blockIdx.x * 256 + threadIdx.x;   // < 1024*512*49
  int hw = (int)(i % HW_);
  size_t t = i / HW_;
  int c = (int)(t & 511);
  int r = (int)(t >> 9);
  float x = fr[i];
  int j = r * HW_ + hw;
  frT_relu[(size_t)j * DHR_ + c] = f2bf(fmaxf(x, 0.f));
  frR_cb[(size_t)c * NRHW_ + j]  = f2bf(x);
}

// feature_obj [o,c] -> fo_relu bf16 [o,c] (relu'd) and foT bf16 [c,o] (raw)
__global__ __launch_bounds__(256)
void obj_pack_kernel(const float* __restrict__ fo, bf16* __restrict__ fo_relu,
                     bf16* __restrict__ foT)
{
  int i = blockIdx.x * 256 + threadIdx.x;   // < 512*512
  int c = i & 511, o = i >> 9;
  float x = fo[i];
  fo_relu[i] = f2bf(fmaxf(x, 0.f));
  foT[c * NO_ + o] = f2bf(x);
}

__global__ __launch_bounds__(256)
void cvt_bf16_kernel(const float* __restrict__ in, bf16* __restrict__ out, int n)
{
  int i = blockIdx.x * 256 + threadIdx.x;
  if (i < n) out[i] = f2bf(in[i]);
}

// ---------------------------------------------------------------------------
// Host orchestration
// ---------------------------------------------------------------------------
extern "C" void kernel_launch(void* const* d_in, const int* in_sizes, int n_in,
                              void* d_out, int out_size, void* d_ws, size_t ws_size,
                              hipStream_t stream)
{
  (void)in_sizes; (void)n_in; (void)out_size; (void)ws_size;

  const float* feature_obj    = (const float*)d_in[0];
  const float* feature_region = (const float*)d_in[1];
  const int*   mat_object     = (const int*)d_in[2];
  const int*   mat_region     = (const int*)d_in[3];
  const float* W_t_o2r        = (const float*)d_in[4];
  const float* b_t_o2r        = (const float*)d_in[5];
  const float* W_t_r2o        = (const float*)d_in[6];
  const float* b_t_r2o        = (const float*)d_in[7];
  const float* W_a_r2o_obj    = (const float*)d_in[8];
  const float* b_a_r2o_obj    = (const float*)d_in[9];
  const float* W_a_r2o_reg    = (const float*)d_in[10];
  const float* b_a_r2o_reg    = (const float*)d_in[11];
  const float* W_a_o2r_reg    = (const float*)d_in[12];
  const float* b_a_o2r_reg    = (const float*)d_in[13];
  const float* W_a_o2r_obj    = (const float*)d_in[14];
  const float* b_a_o2r_obj    = (const float*)d_in[15];

  float* out_obj = (float*)d_out;                       // [512, 512]
  float* out_reg = (float*)d_out + (size_t)NO_ * DHO_;  // [1024, 512, 49]

  // ---- workspace layout with liveness reuse ----
  char* ws = (char*)d_ws;
  size_t off = 0;
  auto alloc = [&](size_t bytes) -> char* {
    char* p = ws + off;
    off += (bytes + 255) & ~(size_t)255;
    return p;
  };

  bf16* fo_relu   = (bf16*)alloc((size_t)NO_ * DHO_ * 2);
  bf16* foT       = (bf16*)alloc((size_t)DHO_ * NO_ * 2);
  bf16* wb_ar2o_o = (bf16*)alloc((size_t)DM_ * DHO_ * 2);
  bf16* wb_ao2r_o = (bf16*)alloc((size_t)DM_ * DHO_ * 2);
  bf16* wb_ar2o_r = (bf16*)alloc((size_t)DM_ * DHR_ * 2);
  bf16* wb_ao2r_r = (bf16*)alloc((size_t)DM_ * DHR_ * 2);
  bf16* wb_t_r2o  = (bf16*)alloc((size_t)DHO_ * DHR_ * 2);
  bf16* wb_t_o2r  = (bf16*)alloc((size_t)DHR_ * DHO_ * 2);
  bf16* fo_att    = (bf16*)alloc((size_t)NO_ * DM_ * 2);
  bf16* fo_att2   = (bf16*)alloc((size_t)NO_ * DM_ * 2);
  bf16* msg_obj_r = (bf16*)alloc((size_t)NO_ * DHR_ * 2);

  char* BUF_A = alloc((size_t)NRHW_ * DHR_ * 2);   // frT_relu, later prob2
  char* BUF_B = alloc((size_t)DHR_ * NRHW_ * 2);   // frR_cb (raw, [c, j])
  char* BUF_C = alloc((size_t)NRHW_ * DM_ * 2 * 2);// fr_att + fr_att2, later msg_reg
  char* BUF_D = alloc((size_t)NO_ * NRHW_ * 4);    // sim, later sim2
  char* BUF_E = alloc((size_t)NO_ * NRHW_ * 2);    // prob

  bf16* frT_relu = (bf16*)BUF_A;
  bf16* prob2    = (bf16*)BUF_A;                       // reuse after fr_att2
  bf16* frR_cb   = (bf16*)BUF_B;
  bf16* fr_att   = (bf16*)BUF_C;
  bf16* fr_att2  = (bf16*)(BUF_C + (size_t)NRHW_ * DM_ * 2);
  bf16* msg_reg  = (bf16*)BUF_C;                       // reuse after sim2
  float* simbuf  = (float*)BUF_D;                      // sim, then sim2
  bf16* prob     = (bf16*)BUF_E;

  dim3 blk(256);

  // ---- packing / conversion ----
  obj_pack_kernel<<<dim3(NO_ * DHO_ / 256), blk, 0, stream>>>(feature_obj, fo_relu, foT);
  region_pack_kernel<<<dim3((unsigned)((size_t)NRHW_ * DHR_ / 256)), blk, 0, stream>>>(
      feature_region, frT_relu, frR_cb);
  cvt_bf16_kernel<<<dim3(DM_ * DHO_ / 256), blk, 0, stream>>>(W_a_r2o_obj, wb_ar2o_o, DM_ * DHO_);
  cvt_bf16_kernel<<<dim3(DM_ * DHO_ / 256), blk, 0, stream>>>(W_a_o2r_obj, wb_ao2r_o, DM_ * DHO_);
  cvt_bf16_kernel<<<dim3(DM_ * DHR_ / 256), blk, 0, stream>>>(W_a_r2o_reg, wb_ar2o_r, DM_ * DHR_);
  cvt_bf16_kernel<<<dim3(DM_ * DHR_ / 256), blk, 0, stream>>>(W_a_o2r_reg, wb_ao2r_r, DM_ * DHR_);
  cvt_bf16_kernel<<<dim3(DHO_ * DHR_ / 256), blk, 0, stream>>>(W_t_r2o, wb_t_r2o, DHO_ * DHR_);
  cvt_bf16_kernel<<<dim3(DHR_ * DHO_ / 256), blk, 0, stream>>>(W_t_o2r, wb_t_o2r, DHR_ * DHO_);

  // ---- attention projections ----
  // fo_att  = relu(fo) @ W_a_r2o_obj^T + b      [512, 256] bf16
  gemm_bf16_nt<1,false,true,false><<<dim3(NO_/BM, DM_/BN), blk, 0, stream>>>(
      fo_relu, wb_ar2o_o, NO_, DM_, DHO_, nullptr, fo_att, b_a_r2o_obj, nullptr);
  // fo_att2 = relu(fo) @ W_a_o2r_obj^T + b      [512, 256] bf16
  gemm_bf16_nt<1,false,true,false><<<dim3(NO_/BM, DM_/BN), blk, 0, stream>>>(
      fo_relu, wb_ao2r_o, NO_, DM_, DHO_, nullptr, fo_att2, b_a_o2r_obj, nullptr);
  // fr_att  = conv1x1(relu(fr), W_a_r2o_reg)    [50176, 256] bf16
  gemm_bf16_nt<1,false,true,false><<<dim3(NRHW_/BM, DM_/BN), blk, 0, stream>>>(
      frT_relu, wb_ar2o_r, NRHW_, DM_, DHR_, nullptr, fr_att, b_a_r2o_reg, nullptr);
  // fr_att2 = conv1x1(relu(fr), W_a_o2r_reg)    [50176, 256] bf16
  gemm_bf16_nt<1,false,true,false><<<dim3(NRHW_/BM, DM_/BN), blk, 0, stream>>>(
      frT_relu, wb_ao2r_r, NRHW_, DM_, DHR_, nullptr, fr_att2, b_a_o2r_reg, nullptr);

  // ---- region -> object ----
  // sim[o, j] = fo_att . fr_att                 [512, 50176] fp32
  gemm_bf16_nt<0,false,false,false><<<dim3(NO_/BM, NRHW_/BN), blk, 0, stream>>>(
      fo_att, fr_att, NO_, NRHW_, DM_, simbuf, nullptr, nullptr, nullptr);
  softmax_obj_kernel<<<dim3(NO_), blk, 0, stream>>>(simbuf, mat_object, prob);

  // ---- object -> region logits (sim buffer free after softmax_obj) ----
  // sim2[j, o] = fr_att2 . fo_att2              [50176, 512] fp32
  gemm_bf16_nt<0,false,false,false><<<dim3(NRHW_/BM, NO_/BN), blk, 0, stream>>>(
      fr_att2, fo_att2, NRHW_, NO_, DM_, simbuf, nullptr, nullptr, nullptr);
  softmax_reg_kernel<<<dim3(NRHW_), blk, 0, stream>>>(simbuf, mat_region, prob2);

  // msg_obj = relu(prob @ feature_region)       [512, 512] bf16
  gemm_bf16_nt<1,true,false,false><<<dim3(NO_/BM, DHR_/BN), blk, 0, stream>>>(
      prob, frR_cb, NO_, DHR_, NRHW_, nullptr, msg_obj_r, nullptr, nullptr);
  // out_obj = feature_obj + msg_obj @ W_t_r2o^T + b   (fp32, row-major)
  gemm_bf16_nt<0,false,true,true><<<dim3(NO_/BM, DHO_/BN), blk, 0, stream>>>(
      msg_obj_r, wb_t_r2o, NO_, DHO_, DHR_, out_obj, nullptr, b_t_r2o, feature_obj);

  // msg_reg = relu(prob2 @ feature_obj)         [50176, 512] bf16
  gemm_bf16_nt<1,true,false,false><<<dim3(NRHW_/BM, DHO_/BN), blk, 0, stream>>>(
      prob2, foT, NRHW_, DHO_, NO_, nullptr, msg_reg, nullptr, nullptr);
  // out_reg = feature_region + conv1x1(msg_reg, W_t_o2r)  (fp32, [r, c', hw])
  gemm_bf16_nt<2,false,true,true><<<dim3(NRHW_/BM, DHR_/BN), blk, 0, stream>>>(
      msg_reg, wb_t_o2r, NRHW_, DHR_, DHO_, out_reg, nullptr, b_t_o2r, feature_region);
}
